// interact_50165218017327
// MI455X (gfx1250) — matrix operand order repaired and checked by reading, not compiled, not run on hardware
//
#include <hip/hip_runtime.h>
#include <hip/hip_bf16.h>

// ---------------------------------------------------------------------------
// DialogueRNN-style scan for MI455X (gfx1250, wave32).
//   Phase 0: init barrier + vdiag = bk @ Wv + bv
//   Phase 1: WMMA bf16 GEMMs for all dialogue-only projections
//              Q      = dlg @ Wq + bq                      [256 x 768]
//              GXU_s* = dlg @ sWih*[:,E:2E]^T              [256 x 2304] x2
//              GXU_g* = dlg @ gWih*[:,2E:3E]^T             [256 x 2304] x2
//            (2 N-tiles per wave, shared A fragment, K-loop unrolled)
//   Phase 2: persistent multi-WGP sequential scan (incremental K/V cache,
//            softmax attention, 2x bidirectional GRU cells) with agent-scope
//            grid barriers. f32 recurrence, float4-vectorized weight streams.
// ---------------------------------------------------------------------------

#define EDIM 768
#define UDIM 256
#define TH3  2304   // 3*EDIM
#define NSPK 8
#define NB   24
#define NTHR 256
#define NTOT (NB * NTHR)

typedef __attribute__((ext_vector_type(16))) __bf16 v16bf;
typedef __attribute__((ext_vector_type(8)))  float  v8f;

// ------------------------------- init --------------------------------------
__global__ void init_kernel(const float* __restrict__ Wv,
                            const float* __restrict__ bk,
                            const float* __restrict__ bv,
                            float* __restrict__ vdiag,
                            unsigned* __restrict__ bar) {
  int tid = threadIdx.x;
  if (tid < 4) bar[tid] = 0u;
  for (int o = tid; o < EDIM; o += blockDim.x) {
    float s = bv[o];
    for (int i = 0; i < EDIM; ++i) s = fmaf(bk[i], Wv[i * EDIM + o], s);
    vdiag[o] = s;
  }
}

// --------------------------- WMMA GEMM (bf16) ------------------------------
// C[M,N] = A[M,K] * B + bias;  A row-major f32 (converted to bf16 on load).
// transB==0: B is row-major [K,N] (element B[k*ldb+n])
// transB==1: B is row-major [N,K] (element B[n*ldb+k])
// One wave computes a 16x32 C strip (two 16x16 tiles sharing the A fragment).
// A/B fragments use contiguous k order (k = 16*(lane/16) + s) in both; C layout per CDNA5 ISA 7.12.2.
__global__ __launch_bounds__(32)
void wmma_gemm_bf16(const float* __restrict__ A, int lda,
                    const float* __restrict__ B, int ldb, int transB,
                    const float* __restrict__ bias,
                    float* __restrict__ C, int ldc,
                    int Npairs, int K) {
  const int bid   = blockIdx.x;
  const int pairN = bid % Npairs;
  const int tileM = bid / Npairs;
  const int lane  = threadIdx.x;
  const int l15   = lane & 15;
  const bool hi   = lane >= 16;

  v8f acc0 = {}, acc1 = {};
  const float* arow = A + (long)(tileM * 16 + l15) * lda;
  const int ncol0 = pairN * 32 + l15;
  const int ncol1 = ncol0 + 16;

#pragma unroll 2
  for (int k0 = 0; k0 < K; k0 += 32) {
    // A fragment: lanes 0-15 hold K 0..15; lanes 16-31 hold K 16..31
    v16bf a, b0, b1;
    const int ka0 = k0 + (hi ? 16 : 0);
    const int ka1 = ka0 + 8;
#pragma unroll
    for (int j = 0; j < 8; ++j) {
      a[j]     = (__bf16)arow[ka0 + j];
      a[j + 8] = (__bf16)arow[ka1 + j];
    }
    // B fragment: lanes 0-15 column n, K k0..k0+15; lanes 16-31 K k0+16..k0+31
    const int kb = k0 + (hi ? 16 : 0);
    if (transB) {
      const float* brow0 = B + (long)ncol0 * ldb + kb;
      const float* brow1 = B + (long)ncol1 * ldb + kb;
#pragma unroll
      for (int j = 0; j < 16; ++j) { b0[j] = (__bf16)brow0[j]; b1[j] = (__bf16)brow1[j]; }
    } else {
#pragma unroll
      for (int j = 0; j < 16; ++j) {
        const float* brow = B + (long)(kb + j) * ldb;
        b0[j] = (__bf16)brow[ncol0];
        b1[j] = (__bf16)brow[ncol1];
      }
    }
    acc0 = __builtin_amdgcn_wmma_f32_16x16x32_bf16(
        false, a, false, b0, (short)0, acc0, false, false);
    acc1 = __builtin_amdgcn_wmma_f32_16x16x32_bf16(
        false, a, false, b1, (short)0, acc1, false, false);
  }
  const float bb0 = bias ? bias[ncol0] : 0.0f;
  const float bb1 = bias ? bias[ncol1] : 0.0f;
  const int r0 = hi ? 8 : 0;  // lanes 0-15: M 0..7, lanes 16-31: M 8..15
#pragma unroll
  for (int r = 0; r < 8; ++r) {
    C[(long)(tileM * 16 + r0 + r) * ldc + ncol0] = acc0[r] + bb0;
    C[(long)(tileM * 16 + r0 + r) * ldc + ncol1] = acc1[r] + bb1;
  }
}

// --------------------------- scan kernel -----------------------------------
struct ScanParams {
  const float* dialogue; const int* speakers;
  const float* Wk; const float* bk; const float* Wv; const float* bv;
  const float* sWih[2]; const float* sWhh[2]; const float* sbih[2]; const float* sbhh[2];
  const float* gWih[2]; const float* gWhh[2]; const float* gbih[2]; const float* gbhh[2];
  const float* h0_global; const float* h0_speaker; const float* att0;
  float* kbuf; float* vbuf; float* Q;
  float* GXU_s[2]; float* GXU_g[2];
  float* vdiag; float* att; float* scores;
  float* gx_s; float* gh_s; float* gx_g; float* gh_g;
  float* sstates; float* hglob; float* so_cur;
  float* out_buf; float* out_so;
  unsigned* bar;
};

__device__ inline void grid_sync(unsigned* bar) {
  __threadfence();
  __syncthreads();
  if (threadIdx.x == 0) {
    unsigned g = __hip_atomic_load(bar + 1, __ATOMIC_RELAXED, __HIP_MEMORY_SCOPE_AGENT);
    unsigned n = __hip_atomic_fetch_add(bar, 1u, __ATOMIC_ACQ_REL, __HIP_MEMORY_SCOPE_AGENT);
    if (n == (unsigned)(NB - 1)) {
      __hip_atomic_store(bar, 0u, __ATOMIC_RELAXED, __HIP_MEMORY_SCOPE_AGENT);
      __hip_atomic_fetch_add(bar + 1, 1u, __ATOMIC_RELEASE, __HIP_MEMORY_SCOPE_AGENT);
    } else {
      while (__hip_atomic_load(bar + 1, __ATOMIC_ACQUIRE, __HIP_MEMORY_SCOPE_AGENT) == g)
        __builtin_amdgcn_s_sleep(1);
    }
  }
  __syncthreads();
  __threadfence();
}

__device__ inline float reduce8(float s) {
  s += __shfl_xor(s, 1, 8);
  s += __shfl_xor(s, 2, 8);
  s += __shfl_xor(s, 4, 8);
  return s;
}

__device__ inline float sigm(float x) { return 1.0f / (1.0f + expf(-x)); }

__device__ inline float dot4acc(const float4 a, const float4 b, float s) {
  s = fmaf(a.x, b.x, s);
  s = fmaf(a.y, b.y, s);
  s = fmaf(a.z, b.z, s);
  s = fmaf(a.w, b.w, s);
  return s;
}

__global__ __launch_bounds__(NTHR)
void scan_kernel(ScanParams p) {
  const int tid  = threadIdx.x;
  const int gtid = blockIdx.x * NTHR + tid;
  const int sub  = tid & 7;
  const int oidx = blockIdx.x * 32 + (tid >> 3);  // 24 blocks * 32 = 768 channels
  const float inv_norm = 0.036084391824352f;      // 1/sqrt(768)

  __shared__ float red[NTHR];
  __shared__ float wsm[UDIM];

  // init recurrent state
  for (int i = gtid; i < NSPK * 2 * EDIM; i += NTOT)
    p.sstates[i] = p.h0_speaker[i % (2 * EDIM)];
  for (int i = gtid; i < 2 * EDIM; i += NTOT) p.hglob[i] = p.h0_global[i];
  grid_sync(p.bar);

  for (int t = 0; t < UDIM; ++t) {
    const int sp = p.speakers[t];
    const float* q = p.Q + (long)t * EDIM;
    const float* u = p.dialogue + (long)t * EDIM;
    const float4* q4 = (const float4*)q;

    if (t == 0) {
      for (int i = gtid; i < EDIM; i += NTOT) p.att[i] = p.att0[i];
      grid_sync(p.bar);
    } else {
      // ---- Stage 1: knew = Wk^T buf[t-1] + bk ; scores j <= t-2 ----
      {
        const float* brow = p.out_buf + (long)(t - 1) * (2 * EDIM);
        float s = 0.0f;
        for (int i = sub; i < 2 * EDIM; i += 8)
          s = fmaf(brow[i], p.Wk[(long)i * EDIM + oidx], s);
        s = reduce8(s);
        if (sub == 0) p.kbuf[(long)(t - 1) * EDIM + oidx] = s + p.bk[oidx];

        for (int item = gtid; item < (t - 1) * 8; item += NTOT) {
          const int j = item >> 3;
          const int sb = item & 7;
          const float4* k4 = (const float4*)(p.kbuf + (long)j * EDIM);
          float sc = 0.0f;
          for (int c = sb; c < EDIM / 4; c += 8) sc = dot4acc(k4[c], q4[c], sc);
          sc = reduce8(sc);
          if (sb == 0) p.scores[j] = sc * inv_norm;
        }
      }
      grid_sync(p.bar);
      // ---- Stage 2: vnew = Wv^T knew + bv ; scores j = t-1, t ----
      {
        const float* krow = p.kbuf + (long)(t - 1) * EDIM;
        float s = 0.0f;
        for (int i = sub; i < EDIM; i += 8)
          s = fmaf(krow[i], p.Wv[(long)i * EDIM + oidx], s);
        s = reduce8(s);
        if (sub == 0) p.vbuf[(long)(t - 1) * EDIM + oidx] = s + p.bv[oidx];

        if (blockIdx.x < 2) {
          const float4* kk4 = (blockIdx.x == 0) ? (const float4*)krow
                                                : (const float4*)p.bk;  // j=t uses k=bk
          float sc = (tid < EDIM / 4) ? dot4acc(kk4[tid], q4[tid], 0.0f) : 0.0f;
          red[tid] = sc; __syncthreads();
          for (int o2 = 128; o2 > 0; o2 >>= 1) {
            if (tid < o2) red[tid] += red[tid + o2];
            __syncthreads();
          }
          if (tid == 0)
            p.scores[(blockIdx.x == 0) ? (t - 1) : t] = red[0] * inv_norm;
        }
      }
      grid_sync(p.bar);
      // ---- Stage 3: softmax + att = sum_j w_j * v_j ----
      {
        const float v = (tid <= t) ? p.scores[tid] : -3.0e38f;
        red[tid] = v; __syncthreads();
        for (int o2 = 128; o2 > 0; o2 >>= 1) {
          if (tid < o2) red[tid] = fmaxf(red[tid], red[tid + o2]);
          __syncthreads();
        }
        const float m = red[0]; __syncthreads();
        const float ev = (tid <= t) ? expf(v - m) : 0.0f;
        red[tid] = ev; __syncthreads();
        for (int o2 = 128; o2 > 0; o2 >>= 1) {
          if (tid < o2) red[tid] += red[tid + o2];
          __syncthreads();
        }
        const float ssum = red[0]; __syncthreads();
        wsm[tid] = ev / ssum; __syncthreads();

        float acc = 0.0f;
        for (int j = sub; j <= t; j += 8) {
          const float vv = (j == t) ? p.vdiag[oidx] : p.vbuf[(long)j * EDIM + oidx];
          acc = fmaf(wsm[j], vv, acc);
        }
        acc = reduce8(acc);
        if (sub == 0) p.att[oidx] = acc;
      }
      grid_sync(p.bar);
    }

    // ---- Stage 4: speaker GRU gate pre-activations (float4 streams) ----
    {
      const float* hsp = p.sstates + (long)sp * 2 * EDIM;
      const float4* att4 = (const float4*)p.att;
      for (int item = gtid; item < 2 * (2 * TH3); item += NTOT) {
        int src, idx;
        if (item < 2 * TH3) { src = 0; idx = item; }
        else                { src = 1; idx = item - 2 * TH3; }
        const int d = idx / TH3, r = idx % TH3;
        if (src == 0) {
          const float4* W4 = (const float4*)(p.sWih[d] + (long)r * (2 * EDIM));
          float s = p.GXU_s[d][(long)t * TH3 + r] + p.sbih[d][r];
          for (int c = 0; c < EDIM / 4; ++c) s = dot4acc(W4[c], att4[c], s);
          p.gx_s[d * TH3 + r] = s;
        } else {
          const float4* W4 = (const float4*)(p.sWhh[d] + (long)r * EDIM);
          const float4* h4 = (const float4*)(hsp + d * EDIM);
          float s = p.sbhh[d][r];
          for (int c = 0; c < EDIM / 4; ++c) s = dot4acc(W4[c], h4[c], s);
          p.gh_s[d * TH3 + r] = s;
        }
      }
    }
    grid_sync(p.bar);
    // ---- Stage 5: speaker gates -> h, so = [hf,hb] + [att,u] ----
    {
      for (int item = gtid; item < 2 * EDIM; item += NTOT) {
        const int d = item / EDIM, i = item % EDIM;
        const float rg = sigm(p.gx_s[d * TH3 + i] + p.gh_s[d * TH3 + i]);
        const float z  = sigm(p.gx_s[d * TH3 + EDIM + i] + p.gh_s[d * TH3 + EDIM + i]);
        const float n  = tanhf(p.gx_s[d * TH3 + 2 * EDIM + i] +
                               rg * p.gh_s[d * TH3 + 2 * EDIM + i]);
        float* hptr = p.sstates + (long)sp * 2 * EDIM + d * EDIM + i;
        const float hold = *hptr;
        const float hnew = (1.0f - z) * n + z * hold;
        *hptr = hnew;
        const float so = hnew + ((d == 0) ? p.att[i] : u[i]);
        p.so_cur[d * EDIM + i] = so;
        p.out_so[(long)t * 2 * EDIM + d * EDIM + i] = so;
      }
    }
    grid_sync(p.bar);
    // ---- Stage 6: global GRU gate pre-activations (float4 streams) ----
    {
      const float4* so4 = (const float4*)p.so_cur;
      for (int item = gtid; item < 2 * (2 * TH3); item += NTOT) {
        int src, idx;
        if (item < 2 * TH3) { src = 0; idx = item; }
        else                { src = 1; idx = item - 2 * TH3; }
        const int d = idx / TH3, r = idx % TH3;
        if (src == 0) {
          const float4* W4 = (const float4*)(p.gWih[d] + (long)r * TH3);
          float s = p.GXU_g[d][(long)t * TH3 + r] + p.gbih[d][r];
          for (int c = 0; c < (2 * EDIM) / 4; ++c) s = dot4acc(W4[c], so4[c], s);
          p.gx_g[d * TH3 + r] = s;
        } else {
          const float4* W4 = (const float4*)(p.gWhh[d] + (long)r * EDIM);
          const float4* h4 = (const float4*)(p.hglob + d * EDIM);
          float s = p.gbhh[d][r];
          for (int c = 0; c < EDIM / 4; ++c) s = dot4acc(W4[c], h4[c], s);
          p.gh_g[d * TH3 + r] = s;
        }
      }
    }
    grid_sync(p.bar);
    // ---- Stage 7: global gates -> hglob, buf[t] ----
    {
      for (int item = gtid; item < 2 * EDIM; item += NTOT) {
        const int d = item / EDIM, i = item % EDIM;
        const float rg = sigm(p.gx_g[d * TH3 + i] + p.gh_g[d * TH3 + i]);
        const float z  = sigm(p.gx_g[d * TH3 + EDIM + i] + p.gh_g[d * TH3 + EDIM + i]);
        const float n  = tanhf(p.gx_g[d * TH3 + 2 * EDIM + i] +
                               rg * p.gh_g[d * TH3 + 2 * EDIM + i]);
        const float hold = p.hglob[d * EDIM + i];
        const float hnew = (1.0f - z) * n + z * hold;
        p.hglob[d * EDIM + i] = hnew;
        p.out_buf[(long)t * 2 * EDIM + d * EDIM + i] = hnew;
      }
    }
    grid_sync(p.bar);
  }
}

// ------------------------------ launch -------------------------------------
extern "C" void kernel_launch(void* const* d_in, const int* in_sizes, int n_in,
                              void* d_out, int out_size, void* d_ws, size_t ws_size,
                              hipStream_t stream) {
  (void)in_sizes; (void)n_in; (void)out_size; (void)ws_size;

  const float* dialogue = (const float*)d_in[0];
  const int*   speakers = (const int*)d_in[1];
  const float* Wq = (const float*)d_in[2];  const float* bq = (const float*)d_in[3];
  const float* Wk = (const float*)d_in[4];  const float* bk = (const float*)d_in[5];
  const float* Wv = (const float*)d_in[6];  const float* bv = (const float*)d_in[7];
  const float* sWih_f = (const float*)d_in[8];  const float* sWhh_f = (const float*)d_in[9];
  const float* sbih_f = (const float*)d_in[10]; const float* sbhh_f = (const float*)d_in[11];
  const float* sWih_b = (const float*)d_in[12]; const float* sWhh_b = (const float*)d_in[13];
  const float* sbih_b = (const float*)d_in[14]; const float* sbhh_b = (const float*)d_in[15];
  const float* gWih_f = (const float*)d_in[16]; const float* gWhh_f = (const float*)d_in[17];
  const float* gbih_f = (const float*)d_in[18]; const float* gbhh_f = (const float*)d_in[19];
  const float* gWih_b = (const float*)d_in[20]; const float* gWhh_b = (const float*)d_in[21];
  const float* gbih_b = (const float*)d_in[22]; const float* gbhh_b = (const float*)d_in[23];
  const float* h0_global  = (const float*)d_in[24];
  const float* h0_speaker = (const float*)d_in[25];
  const float* att0       = (const float*)d_in[26];

  // workspace carve-out (~12 MB of f32); all offsets 16B-aligned
  float* ws = (float*)d_ws;
  size_t off = 0;
  auto carve = [&](size_t n) { float* r = ws + off; off += n; return r; };
  float* kbuf    = carve((size_t)UDIM * EDIM);
  float* vbuf    = carve((size_t)UDIM * EDIM);
  float* Qbuf    = carve((size_t)UDIM * EDIM);
  float* GXU_sf  = carve((size_t)UDIM * TH3);
  float* GXU_sb  = carve((size_t)UDIM * TH3);
  float* GXU_gf  = carve((size_t)UDIM * TH3);
  float* GXU_gb  = carve((size_t)UDIM * TH3);
  float* vdiag   = carve(EDIM);
  float* att     = carve(EDIM);
  float* scores  = carve(UDIM);
  float* gx_s    = carve(2 * TH3);
  float* gh_s    = carve(2 * TH3);
  float* gx_g    = carve(2 * TH3);
  float* gh_g    = carve(2 * TH3);
  float* sstates = carve((size_t)NSPK * 2 * EDIM);
  float* hglob   = carve(2 * EDIM);
  float* so_cur  = carve(2 * EDIM);
  unsigned* bar  = (unsigned*)(ws + off); off += 16;

  float* out_buf = (float*)d_out;                      // [U, 2E] global_output
  float* out_so  = out_buf + (size_t)UDIM * 2 * EDIM;  // [U, 2E] speaker_output

  // Phase 0
  init_kernel<<<1, 256, 0, stream>>>(Wv, bk, bv, vdiag, bar);

  // Phase 1: WMMA bf16 GEMMs (M=256 -> 16 M-tiles; waves own 16x32 C strips)
  wmma_gemm_bf16<<<16 * (EDIM / 32), 32, 0, stream>>>(
      dialogue, EDIM, Wq, EDIM, 0, bq, Qbuf, EDIM, EDIM / 32, EDIM);
  wmma_gemm_bf16<<<16 * (TH3 / 32), 32, 0, stream>>>(
      dialogue, EDIM, sWih_f + EDIM, 2 * EDIM, 1, nullptr, GXU_sf, TH3, TH3 / 32, EDIM);
  wmma_gemm_bf16<<<16 * (TH3 / 32), 32, 0, stream>>>(
      dialogue, EDIM, sWih_b + EDIM, 2 * EDIM, 1, nullptr, GXU_sb, TH3, TH3 / 32, EDIM);
  wmma_gemm_bf16<<<16 * (TH3 / 32), 32, 0, stream>>>(
      dialogue, EDIM, gWih_f + 2 * EDIM, TH3, 1, nullptr, GXU_gf, TH3, TH3 / 32, EDIM);
  wmma_gemm_bf16<<<16 * (TH3 / 32), 32, 0, stream>>>(
      dialogue, EDIM, gWih_b + 2 * EDIM, TH3, 1, nullptr, GXU_gb, TH3, TH3 / 32, EDIM);

  // Phase 2: persistent sequential scan
  ScanParams p;
  p.dialogue = dialogue; p.speakers = speakers;
  p.Wk = Wk; p.bk = bk; p.Wv = Wv; p.bv = bv;
  p.sWih[0] = sWih_f; p.sWih[1] = sWih_b;
  p.sWhh[0] = sWhh_f; p.sWhh[1] = sWhh_b;
  p.sbih[0] = sbih_f; p.sbih[1] = sbih_b;
  p.sbhh[0] = sbhh_f; p.sbhh[1] = sbhh_b;
  p.gWih[0] = gWih_f; p.gWih[1] = gWih_b;
  p.gWhh[0] = gWhh_f; p.gWhh[1] = gWhh_b;
  p.gbih[0] = gbih_f; p.gbih[1] = gbih_b;
  p.gbhh[0] = gbhh_f; p.gbhh[1] = gbhh_b;
  p.h0_global = h0_global; p.h0_speaker = h0_speaker; p.att0 = att0;
  p.kbuf = kbuf; p.vbuf = vbuf; p.Q = Qbuf;
  p.GXU_s[0] = GXU_sf; p.GXU_s[1] = GXU_sb;
  p.GXU_g[0] = GXU_gf; p.GXU_g[1] = GXU_gb;
  p.vdiag = vdiag; p.att = att; p.scores = scores;
  p.gx_s = gx_s; p.gh_s = gh_s; p.gx_g = gx_g; p.gh_g = gh_g;
  p.sstates = sstates; p.hglob = hglob; p.so_cur = so_cur;
  p.out_buf = out_buf; p.out_so = out_so; p.bar = bar;

  scan_kernel<<<NB, NTHR, 0, stream>>>(p);
}
